// PRNNModel_11991548690511
// MI455X (gfx1250) — matrix operand order repaired and checked by reading, not compiled, not run on hardware
//
#include <hip/hip_runtime.h>
#include <hip/hip_bf16.h>
#include <stdint.h>

#define DEV __device__ __forceinline__

typedef __attribute__((ext_vector_type(16))) __bf16 v16bf;
typedef __attribute__((ext_vector_type(8)))  __bf16 v8bf;
typedef __attribute__((ext_vector_type(8)))  float  v8f;

// Problem constants (fixed by the reference)
constexpr int kB  = 64;
constexpr int kT  = 512;
constexpr int kE  = 512;
constexpr int kH  = 1024;
constexpr int kD  = 1024;   // 2E, GRU input width; also K of every GEMM here
constexpr int kG3 = 3072;   // 3H
constexpr int kV0 = 2048;
constexpr int kV1 = 128;

// ---------------------------------------------------------------------------
// WMMA fragment helpers (ISA 7.12.2 layouts, wave32)
// ---------------------------------------------------------------------------

DEV v8f wmma_bf16(v16bf a, v16bf b, v8f c) {
  return __builtin_amdgcn_wmma_f32_16x16x32_bf16(false, a, false, b,
                                                 (short)0, c, false, false);
}

// A-matrix 16x32 bf16 fragment: lane l holds row m=l&15;
// lanes 0-15: K = {0..7, 16..23}; lanes 16-31: K = {8..15, 24..31}.
DEV v16bf load_a16(const __bf16* __restrict__ base /*row m0, col k0*/,
                   int ld, int lane) {
  const int m  = lane & 15;
  const int ko = (lane < 16) ? 0 : 8;
  const __bf16* p = base + (size_t)m * ld + ko;
  v8bf c0 = *(const v8bf*)(p);
  v8bf c1 = *(const v8bf*)(p + 16);
  return __builtin_shufflevector(c0, c1, 0,1,2,3,4,5,6,7,8,9,10,11,12,13,14,15);
}

// A fragment from a per-lane row pointer (conditional history gather).
DEV v16bf load_a16_row(const __bf16* __restrict__ rowp, int k0, int lane) {
  const int ko = k0 + ((lane < 16) ? 0 : 8);
  v8bf c0 = *(const v8bf*)(rowp + ko);
  v8bf c1 = *(const v8bf*)(rowp + ko + 16);
  return __builtin_shufflevector(c0, c1, 0,1,2,3,4,5,6,7,8,9,10,11,12,13,14,15);
}

// B-matrix 32x16 bf16 fragment from row-major weights W[N][K]:
// lane l holds weight row n=n0+(l&15);
// lanes 0-15: K = {0..7, 16..23}; lanes 16-31: K = {8..15, 24..31}.
DEV v16bf load_b16(const __bf16* __restrict__ base /*row n0, col k0*/,
                   int ld, int lane) {
  const int n  = lane & 15;
  const int ko = (lane < 16) ? 0 : 8;
  const __bf16* p = base + (size_t)n * ld + ko;
  v8bf c0 = *(const v8bf*)(p);
  v8bf c1 = *(const v8bf*)(p + 16);
  return __builtin_shufflevector(c0, c1, 0,1,2,3,4,5,6,7,8,9,10,11,12,13,14,15);
}

DEV v8f bcast8(float x) { v8f r = {x,x,x,x,x,x,x,x}; return r; }

// C/D layout: lane l, acc[v] -> (m = m0 + v + 8*(l>=16), n = n0 + (l&15))
DEV void store_acc_f32(float* __restrict__ C, int ldc, int m0, int n0,
                       const v8f& acc, int lane) {
  const int n  = n0 + (lane & 15);
  const int mb = m0 + ((lane >> 4) << 3);
#pragma unroll
  for (int v = 0; v < 8; ++v) C[(size_t)(mb + v) * ldc + n] = acc[v];
}

// Generic 16(M) x (NT*16)(N) wave tile, K = 1024, A-fragment reused NT times.
// Software-pipelined and FULLY UNROLLED: fragments for K-step k+1 are issued
// before the WMMAs of step k; the unroll lets register renaming absorb the
// ping-pong rotation (no v_mov walls) while the backend emits partial
// s_wait_loadcnt to hide load latency behind WMMA.
template <int NT>
DEV void wmma_rowtile(const __bf16* __restrict__ A, int m0,
                      const __bf16* __restrict__ W, int n0,
                      const float* __restrict__ bias, v8f* acc, int lane) {
  const int n = lane & 15;
#pragma unroll
  for (int j = 0; j < NT; ++j)
    acc[j] = bcast8(bias ? bias[n0 + j * 16 + n] : 0.f);

  const __bf16* Ab = A + (size_t)m0 * kD;
  v16bf af = load_a16(Ab, kD, lane);
  v16bf bf[NT];
#pragma unroll
  for (int j = 0; j < NT; ++j)
    bf[j] = load_b16(W + (size_t)(n0 + j * 16) * kD, kD, lane);

#pragma unroll
  for (int k0 = 32; k0 < kD; k0 += 32) {
    v16bf afn = load_a16(Ab + k0, kD, lane);
    v16bf bfn[NT];
#pragma unroll
    for (int j = 0; j < NT; ++j)
      bfn[j] = load_b16(W + (size_t)(n0 + j * 16) * kD + k0, kD, lane);
#pragma unroll
    for (int j = 0; j < NT; ++j) acc[j] = wmma_bf16(af, bf[j], acc[j]);
    af = afn;
#pragma unroll
    for (int j = 0; j < NT; ++j) bf[j] = bfn[j];
  }
#pragma unroll
  for (int j = 0; j < NT; ++j) acc[j] = wmma_bf16(af, bf[j], acc[j]);
}

DEV float sigm(float x) { return 1.f / (1.f + __expf(-x)); }

// ---------------------------------------------------------------------------
// Setup kernels
// ---------------------------------------------------------------------------

__global__ void f32_to_bf16(const float* __restrict__ s, __bf16* __restrict__ d,
                            int n) {
  int i = blockIdx.x * 256 + threadIdx.x;
  if (i < n) d[i] = (__bf16)s[i];
}

__global__ void zero_u32(uint32_t* __restrict__ p, int n) {
  int i = blockIdx.x * 256 + threadIdx.x;
  if (i < n) p[i] = 0u;
}

// X[t][b][e] = bf16(concat(emb0[data0[b,t]], emb1[data1[b,t]])[e])
__global__ void embed_concat(const int* __restrict__ d0,
                             const int* __restrict__ d1,
                             const float* __restrict__ e0,
                             const float* __restrict__ e1,
                             __bf16* __restrict__ X) {
  int idx = blockIdx.x * 256 + threadIdx.x;   // T*B*D threads exactly
  int t = idx >> 16;          // / (B*D) = 65536
  int r = idx & 65535;
  int b = r >> 10;
  int e = r & 1023;
  float v;
  if (e < kE) v = e0[(size_t)d0[b * kT + t] * kE + e];
  else        v = e1[(size_t)d1[b * kT + t] * kE + (e - kE)];
  X[(size_t)idx] = (__bf16)v;
}

// ---------------------------------------------------------------------------
// Big parallel GEMMs
// ---------------------------------------------------------------------------

// Gi[T*B, 3H] = X[T*B, 1024] @ W_ih^T + b_ih    (hoisted out of the scan:
// both GRU cells share gi, and it has no recurrent dependence)
__global__ __launch_bounds__(256) void gi_gemm(const __bf16* __restrict__ X,
                                               const __bf16* __restrict__ Wih,
                                               const float* __restrict__ bih,
                                               float* __restrict__ Gi) {
  const int lane = threadIdx.x & 31, w = threadIdx.x >> 5;
  const int m0 = blockIdx.x * 32 + (w & 1) * 16;
  const int n0 = blockIdx.y * 256 + (w >> 1) * 64;
  v8f acc[4];
  wmma_rowtile<4>(X, m0, Wih, n0, bih, acc, lane);
#pragma unroll
  for (int j = 0; j < 4; ++j)
    store_acc_f32(Gi, kG3, m0, n0 + j * 16, acc[j], lane);
}

// dec[b][t][v] = Out[t*64+b, :] @ Wd^T + bd   (writes reference [B,T,V] layout)
__global__ __launch_bounds__(256) void dec_gemm(const __bf16* __restrict__ Out,
                                                const __bf16* __restrict__ Wd,
                                                const float* __restrict__ bd,
                                                float* __restrict__ dst, int V) {
  const int lane = threadIdx.x & 31, w = threadIdx.x >> 5;
  const int m0 = blockIdx.x * 32 + (w & 1) * 16;
  const int n0 = blockIdx.y * 256 + (w >> 1) * 64;
  if (n0 >= V) return;
  v8f acc[4];
  wmma_rowtile<4>(Out, m0, Wd, n0, bd, acc, lane);
  const int mb = m0 + ((lane >> 4) << 3);
#pragma unroll
  for (int j = 0; j < 4; ++j) {
    const int n = n0 + j * 16 + (lane & 15);
#pragma unroll
    for (int v = 0; v < 8; ++v) {
      int row = mb + v, t = row >> 6, b = row & 63;
      dst[(size_t)b * kT * V + (size_t)t * V + n] = acc[j][v];
    }
  }
}

// ---------------------------------------------------------------------------
// Per-timestep kernels (sequential chain; weights stay hot in 192MB L2)
// ---------------------------------------------------------------------------

// stepA: blocks [0,32):   nh  = main_h@A^T + select(prev)@Bmat^T   [64,1024]
//        blocks [32,128): ghp = par_h@W_hh^T + b_hh                [64,3072]
// History `prevs` is kept in bf16 (it is only ever a GEMM A-operand), so the
// gather is two plain 16B vector loads per lane — no per-element converts.
__global__ __launch_bounds__(256) void stepA(
    const __bf16* __restrict__ mainbf, const __bf16* __restrict__ parbf,
    const __bf16* __restrict__ Abf, const __bf16* __restrict__ Bmbf,
    const __bf16* __restrict__ Whh, const __bf16* __restrict__ prevsbf,
    const __bf16* __restrict__ defhbf, const int* __restrict__ cond,
    const float* __restrict__ bhh, float* __restrict__ nhf,
    __bf16* __restrict__ nhbf, float* __restrict__ ghp, int t) {
  const int lane = threadIdx.x & 31, w = threadIdx.x >> 5;
  const int m0 = (w & 3) * 16;
  const int ctl = w >> 2;

  if (blockIdx.x < 32) {
    const int n0 = blockIdx.x * 32 + ctl * 16;
    // per-lane conditional history gather (reference get_new_h_t semantics)
    const int b = m0 + (lane & 15);
    const int ct = cond[b * kT + t];
    const bool use = (t > 2) && (ct != -1) && (ct < t - 1);   // SKIP=2, W_OFF=1
    int idx = ct + 1;                                          // + W_OFF
    idx = idx < 0 ? 0 : (idx > kT ? kT : idx);
    const __bf16* prow =
        use ? (prevsbf + (size_t)idx * kB * kH + (size_t)b * kH) : defhbf;

    const __bf16* Am = mainbf + (size_t)m0 * kD;
    const __bf16* W1 = Abf + (size_t)n0 * kD;
    const __bf16* W2 = Bmbf + (size_t)n0 * kD;

    v8f acc = bcast8(0.f);
    // software-pipelined: prefetch k+1 fragments before the k WMMAs
    v16bf a1 = load_a16(Am, kD, lane);
    v16bf b1 = load_b16(W1, kD, lane);
    v16bf a2 = load_a16_row(prow, 0, lane);
    v16bf b2 = load_b16(W2, kD, lane);
#pragma unroll
    for (int k0 = 32; k0 < kD; k0 += 32) {
      v16bf a1n = load_a16(Am + k0, kD, lane);
      v16bf b1n = load_b16(W1 + k0, kD, lane);
      v16bf a2n = load_a16_row(prow, k0, lane);
      v16bf b2n = load_b16(W2 + k0, kD, lane);
      acc = wmma_bf16(a1, b1, acc);
      acc = wmma_bf16(a2, b2, acc);
      a1 = a1n; b1 = b1n; a2 = a2n; b2 = b2n;
    }
    acc = wmma_bf16(a1, b1, acc);
    acc = wmma_bf16(a2, b2, acc);

    const int n = n0 + (lane & 15);
    const int mb = m0 + ((lane >> 4) << 3);
#pragma unroll
    for (int v = 0; v < 8; ++v) {
      float x = acc[v];
      nhf[(mb + v) * kH + n] = x;
      nhbf[(mb + v) * kH + n] = (__bf16)x;
    }
  } else {
    const int n0 = (blockIdx.x - 32) * 32 + ctl * 16;
    v8f acc[1];
    wmma_rowtile<1>(parbf, m0, Whh, n0, bhh, acc, lane);
    store_acc_f32(ghp, kG3, m0, n0, acc[0], lane);
  }
}

// stepB: gh_main = nh@W_hh^T + b_hh (r/z/n gate tiles share one A fragment),
// then the full GRU elementwise for BOTH cells fused in registers, plus the
// prevs[t+1] append. 16 blocks x 512 threads cover [64 x 1024] outputs.
__global__ __launch_bounds__(512) void stepB(
    const __bf16* __restrict__ nhbf, const __bf16* __restrict__ Whh,
    const float* __restrict__ bhh, const float* __restrict__ Gi,
    const float* __restrict__ nhf, const float* __restrict__ ghp,
    float* __restrict__ parf, __bf16* __restrict__ prevsbf,
    __bf16* __restrict__ outbf, __bf16* __restrict__ mainbf,
    __bf16* __restrict__ parbf, int t) {
  const int lane = threadIdx.x & 31, w = threadIdx.x >> 5;
  const int m0 = (w & 3) * 16;
  const int hc = blockIdx.x * 64 + (w >> 2) * 16;   // h column-tile base
  const int nn = lane & 15;

  const __bf16* An = nhbf + (size_t)m0 * kD;
  const __bf16* Wr = Whh + (size_t)(hc)*kD;
  const __bf16* Wz = Whh + (size_t)(kH + hc) * kD;
  const __bf16* Wn = Whh + (size_t)(2 * kH + hc) * kD;

  v8f ar = bcast8(bhh[hc + nn]);
  v8f az = bcast8(bhh[kH + hc + nn]);
  v8f an = bcast8(bhh[2 * kH + hc + nn]);

  v16bf af = load_a16(An, kD, lane);
  v16bf br = load_b16(Wr, kD, lane);
  v16bf bz = load_b16(Wz, kD, lane);
  v16bf bn = load_b16(Wn, kD, lane);
#pragma unroll
  for (int k0 = 32; k0 < kD; k0 += 32) {
    v16bf afn = load_a16(An + k0, kD, lane);
    v16bf brn = load_b16(Wr + k0, kD, lane);
    v16bf bzn = load_b16(Wz + k0, kD, lane);
    v16bf bnn = load_b16(Wn + k0, kD, lane);
    ar = wmma_bf16(af, br, ar);
    az = wmma_bf16(af, bz, az);
    an = wmma_bf16(af, bn, an);
    af = afn; br = brn; bz = bzn; bn = bnn;
  }
  ar = wmma_bf16(af, br, ar);
  az = wmma_bf16(af, bz, az);
  an = wmma_bf16(af, bn, an);

  const int h = hc + nn;
  const int mb = m0 + ((lane >> 4) << 3);
  const float* git = Gi + (size_t)t * kB * kG3;
#pragma unroll
  for (int v = 0; v < 8; ++v) {
    const int b = mb + v;
    const float gr = git[b * kG3 + h];
    const float gz = git[b * kG3 + kH + h];
    const float gn = git[b * kG3 + 2 * kH + h];
    // main cell: hidden input is new_h_t (nh)
    float r = sigm(gr + ar[v]);
    float z = sigm(gz + az[v]);
    float n = tanhf(gn + r * an[v]);
    float nh = nhf[b * kH + h];
    float newm = (1.f - z) * n + z * nh;
    outbf[(size_t)t * kB * kH + b * kH + h] = (__bf16)newm;
    mainbf[b * kH + h] = (__bf16)newm;
    // parallel cell (same weights), and append pre-update h to history
    float pr = sigm(gr + ghp[b * kG3 + h]);
    float pz = sigm(gz + ghp[b * kG3 + kH + h]);
    float pn = tanhf(gn + pr * ghp[b * kG3 + 2 * kH + h]);
    float ph = parf[b * kH + h];
    prevsbf[(size_t)(t + 1) * kB * kH + b * kH + h] = (__bf16)ph;
    float newp = (1.f - pz) * pn + pz * ph;
    parf[b * kH + h] = newp;
    parbf[b * kH + h] = (__bf16)newp;
  }
}

// ---------------------------------------------------------------------------
// Host launcher
// ---------------------------------------------------------------------------

static inline char* carve(char*& p, size_t bytes) {
  char* r = p;
  p += (bytes + 255) & ~(size_t)255;
  return r;
}

extern "C" void kernel_launch(void* const* d_in, const int* in_sizes, int n_in,
                              void* d_out, int out_size, void* d_ws,
                              size_t ws_size, hipStream_t stream) {
  const int*   data0 = (const int*)d_in[0];
  const int*   data1 = (const int*)d_in[1];
  const int*   cond  = (const int*)d_in[2];
  const float* emb0  = (const float*)d_in[3];
  const float* emb1  = (const float*)d_in[4];
  const float* Wih   = (const float*)d_in[5];
  const float* Whh   = (const float*)d_in[6];
  const float* bih   = (const float*)d_in[7];
  const float* bhh   = (const float*)d_in[8];
  const float* Amat  = (const float*)d_in[9];
  const float* Bmat  = (const float*)d_in[10];
  const float* defh  = (const float*)d_in[11];
  const float* Wd0   = (const float*)d_in[12];
  const float* bd0   = (const float*)d_in[13];
  const float* Wd1   = (const float*)d_in[14];
  const float* bd1   = (const float*)d_in[15];
  float* out = (float*)d_out;

  // Workspace carve (deterministic, recomputed every call)
  char* p = (char*)d_ws;
  __bf16* Wih_bf  = (__bf16*)carve(p, (size_t)kG3 * kD * 2);
  __bf16* Whh_bf  = (__bf16*)carve(p, (size_t)kG3 * kH * 2);
  __bf16* A_bf    = (__bf16*)carve(p, (size_t)kH * kH * 2);
  __bf16* Bm_bf   = (__bf16*)carve(p, (size_t)kH * kH * 2);
  __bf16* Wd0_bf  = (__bf16*)carve(p, (size_t)kV0 * kH * 2);
  __bf16* Wd1_bf  = (__bf16*)carve(p, (size_t)kV1 * kH * 2);
  __bf16* defh_bf = (__bf16*)carve(p, (size_t)kH * 2);
  __bf16* Xbf     = (__bf16*)carve(p, (size_t)kT * kB * kD * 2);
  float*  Gi      = (float*)carve(p, (size_t)kT * kB * kG3 * 4);
  __bf16* prevsbf = (__bf16*)carve(p, (size_t)(kT + 1) * kB * kH * 2);
  __bf16* Out_bf  = (__bf16*)carve(p, (size_t)kT * kB * kH * 2);
  float*  nhf     = (float*)carve(p, (size_t)kB * kH * 4);
  __bf16* nhbf    = (__bf16*)carve(p, (size_t)kB * kH * 2);
  float*  ghp     = (float*)carve(p, (size_t)kB * kG3 * 4);
  __bf16* mainbf  = (__bf16*)carve(p, (size_t)kB * kH * 2);
  float*  parf    = (float*)carve(p, (size_t)kB * kH * 4);
  __bf16* parbf   = (__bf16*)carve(p, (size_t)kB * kH * 2);

  auto cvt = [&](const float* s, __bf16* d, int n) {
    f32_to_bf16<<<(n + 255) / 256, 256, 0, stream>>>(s, d, n);
  };
  cvt(Wih, Wih_bf, kG3 * kD);
  cvt(Whh, Whh_bf, kG3 * kH);
  cvt(Amat, A_bf, kH * kH);
  cvt(Bmat, Bm_bf, kH * kH);
  cvt(Wd0, Wd0_bf, kV0 * kH);
  cvt(Wd1, Wd1_bf, kV1 * kH);
  cvt(defh, defh_bf, kH);

  // Zero recurrent state + prevs[0]
  zero_u32<<<(kB * kH * 2 / 4 + 255) / 256, 256, 0, stream>>>(
      (uint32_t*)mainbf, kB * kH * 2 / 4);
  zero_u32<<<(kB * kH * 2 / 4 + 255) / 256, 256, 0, stream>>>(
      (uint32_t*)parbf, kB * kH * 2 / 4);
  zero_u32<<<(kB * kH + 255) / 256, 256, 0, stream>>>((uint32_t*)parf,
                                                      kB * kH);
  zero_u32<<<(kB * kH * 2 / 4 + 255) / 256, 256, 0, stream>>>(
      (uint32_t*)prevsbf, kB * kH * 2 / 4);

  // Embedding gather + concat into bf16 [T,B,2E]
  embed_concat<<<(kT * kB * kD) / 256, 256, 0, stream>>>(data0, data1, emb0,
                                                         emb1, Xbf);

  // Hoisted input-gate GEMM: Gi = X @ W_ih^T + b_ih  [T*B, 3H]
  gi_gemm<<<dim3((kT * kB) / 32, kG3 / 256), 256, 0, stream>>>(Xbf, Wih_bf,
                                                               bih, Gi);

  // Sequential recurrence
  for (int t = 0; t < kT; ++t) {
    stepA<<<128, 256, 0, stream>>>(mainbf, parbf, A_bf, Bm_bf, Whh_bf, prevsbf,
                                   defh_bf, cond, bhh, nhf, nhbf, ghp, t);
    stepB<<<16, 512, 0, stream>>>(nhbf, Whh_bf, bhh, Gi, nhf, ghp, parf,
                                  prevsbf, Out_bf, mainbf, parbf, t);
  }

  // Decoders straight into d_out (dec0 then dec1, reference [B,T,V] layout)
  dec_gemm<<<dim3((kT * kB) / 32, kV0 / 256), 256, 0, stream>>>(
      Out_bf, Wd0_bf, bd0, out, kV0);
  dec_gemm<<<dim3((kT * kB) / 32, 1), 256, 0, stream>>>(
      Out_bf, Wd1_bf, bd1, out + (size_t)kB * kT * kV0, kV1);
}
